// ScaledDotProductAttention_49143015801006
// MI455X (gfx1250) — matrix operand order repaired and checked by reading, not compiled, not run on hardware
//
#include <hip/hip_runtime.h>
#include <cstdint>

// ---------------------------------------------------------------------------
// Scaled dot-product attention (transposed layout) for MI455X / gfx1250.
//   scores[b,h,s,t] = sum_d K[b,h,d,s]*Q[b,h,d,t] / 8 ; mask on t; softmax on t
//   ctx[b,t,h*64+d] = sum_s V[b,h,d,s]*attn[b,h,s,t]
// Kernel 1: QK^T + masked softmax -> attn (f32 out, f16 WMMA, f32 accum),
//           double-buffered per-wave Q staging overlapped with WMMA, all
//           staging stores packed (b64).
// Kernel 2: V @ attn -> ctx, block-level double-buffered staging, packed
//           b64/b128 DS stores, LDS transpose epilogue for coalesced output.
// ---------------------------------------------------------------------------

typedef __attribute__((ext_vector_type(16))) _Float16 v16h;
typedef __attribute__((ext_vector_type(8)))  float    v8f;

#define SEQ 1024
#define DH  64
#define NHEAD 12
#define NBATCH 8
#define CTX_ELEMS ((size_t)NBATCH * SEQ * NHEAD * DH)   // 6,291,456

union V16 {
  uint4 u[2];
  v16h  v;
};
union Pack4 { _Float16 h[4]; uint2 u; };
union Pack8 { _Float16 h[8]; uint4 u; };

// Intra-wave LDS producer->consumer fence (cross-lane data through LDS).
__device__ __forceinline__ void lds_fence_wave() {
  __builtin_amdgcn_wave_barrier();
  asm volatile("s_wait_dscnt 0" ::: "memory");
  __builtin_amdgcn_wave_barrier();
}

// ===========================================================================
// Kernel 1: scores = K^T Q (scaled, masked) + softmax over t, write attn f32.
// Block: 256 threads (8 waves). Block handles 16 s-rows; wave handles 128 t.
// ===========================================================================
__global__ __launch_bounds__(256)
void attn_qk_softmax_kernel(const float* __restrict__ Qg,
                            const float* __restrict__ Kg,
                            const int*   __restrict__ maskg,
                            float*       __restrict__ attn)
{
  __shared__ _Float16 ldsA[16][72];          // K^T tile [s][d]  (144B rows)
  __shared__ _Float16 ldsB[8][2][16][72];    // per-wave double-buffered Q [t][d]
  __shared__ float    ldsRed[2][8][16];      // cross-wave row max / row sum

  const int bh = blockIdx.y;
  const int b  = bh / NHEAD;
  const int s0 = blockIdx.x << 4;

  const float* Kp = Kg + (size_t)bh * DH * SEQ;
  const float* Qp = Qg + (size_t)bh * DH * SEQ;
  const int*   mp = maskg + b * SEQ;

  const int tid  = threadIdx.x;
  const int wave = tid >> 5;
  const int lane = tid & 31;
  const int lo16 = lane & 15;
  const int hi   = lane >> 4;

  // ---- stage K^T tile: ldsA[m][d0..d0+3] as one packed b64 store ----
  {
    const int m  = tid & 15;
    const int d0 = (tid >> 4) << 2;
    const float* kp = Kp + s0 + m;
    Pack4 pk;
    pk.h[0] = (_Float16)kp[(size_t)(d0 + 0) * SEQ];
    pk.h[1] = (_Float16)kp[(size_t)(d0 + 1) * SEQ];
    pk.h[2] = (_Float16)kp[(size_t)(d0 + 2) * SEQ];
    pk.h[3] = (_Float16)kp[(size_t)(d0 + 3) * SEQ];
    *reinterpret_cast<uint2*>(&ldsA[m][d0]) = pk.u;
  }
  __syncthreads();

  // A operands (16x32 f16 layout)
  V16 aLo, aHi;
  aLo.u[0] = *reinterpret_cast<const uint4*>(&ldsA[lo16][hi * 8]);
  aLo.u[1] = *reinterpret_cast<const uint4*>(&ldsA[lo16][16 + hi * 8]);
  aHi.u[0] = *reinterpret_cast<const uint4*>(&ldsA[lo16][32 + hi * 8]);
  aHi.u[1] = *reinterpret_cast<const uint4*>(&ldsA[lo16][48 + hi * 8]);

  // Q staging: lane owns 4 t-cols (tg) x 4 consecutive d-rows (db), twice (q).
  const int tg = (lane & 3) << 2;   // 0,4,8,12
  const int db = (lane >> 2) << 2;  // 0,4,...,28

  float4 qreg[8];                   // [q*4 + j]: row q*32+db+j, cols tg..tg+3
  auto loadQ = [&](int c) {
    const int t0 = (wave << 7) + (c << 4);
    #pragma unroll
    for (int q = 0; q < 2; ++q)
      #pragma unroll
      for (int j = 0; j < 4; ++j)
        qreg[q * 4 + j] = *reinterpret_cast<const float4*>(
            Qp + (size_t)(q * 32 + db + j) * SEQ + t0 + tg);
  };
  auto storeQ = [&](int buf) {
    #pragma unroll
    for (int q = 0; q < 2; ++q) {
      const int dbase = q * 32 + db;
      Pack4 p0, p1, p2, p3;   // 4x4 in-register transpose -> d-contiguous
      p0.h[0] = (_Float16)qreg[q*4+0].x; p0.h[1] = (_Float16)qreg[q*4+1].x;
      p0.h[2] = (_Float16)qreg[q*4+2].x; p0.h[3] = (_Float16)qreg[q*4+3].x;
      p1.h[0] = (_Float16)qreg[q*4+0].y; p1.h[1] = (_Float16)qreg[q*4+1].y;
      p1.h[2] = (_Float16)qreg[q*4+2].y; p1.h[3] = (_Float16)qreg[q*4+3].y;
      p2.h[0] = (_Float16)qreg[q*4+0].z; p2.h[1] = (_Float16)qreg[q*4+1].z;
      p2.h[2] = (_Float16)qreg[q*4+2].z; p2.h[3] = (_Float16)qreg[q*4+3].z;
      p3.h[0] = (_Float16)qreg[q*4+0].w; p3.h[1] = (_Float16)qreg[q*4+1].w;
      p3.h[2] = (_Float16)qreg[q*4+2].w; p3.h[3] = (_Float16)qreg[q*4+3].w;
      *reinterpret_cast<uint2*>(&ldsB[wave][buf][tg + 0][dbase]) = p0.u;
      *reinterpret_cast<uint2*>(&ldsB[wave][buf][tg + 1][dbase]) = p1.u;
      *reinterpret_cast<uint2*>(&ldsB[wave][buf][tg + 2][dbase]) = p2.u;
      *reinterpret_cast<uint2*>(&ldsB[wave][buf][tg + 3][dbase]) = p3.u;
    }
  };

  // prologue: chunk 0 into buffer 0
  loadQ(0);
  storeQ(0);

  v8f sc[8];   // live score tiles: cols wave*128+c*16+lo16, rows hi*8+r

  #pragma unroll
  for (int c = 0; c < 8; ++c) {
    const int buf = c & 1;
    if (c < 7) loadQ(c + 1);          // issue next chunk's global loads early
    lds_fence_wave();                 // chunk c's DS stores complete

    V16 bLo, bHi;
    bLo.u[0] = *reinterpret_cast<const uint4*>(&ldsB[wave][buf][lo16][hi * 8]);
    bLo.u[1] = *reinterpret_cast<const uint4*>(&ldsB[wave][buf][lo16][16 + hi * 8]);
    bHi.u[0] = *reinterpret_cast<const uint4*>(&ldsB[wave][buf][lo16][32 + hi * 8]);
    bHi.u[1] = *reinterpret_cast<const uint4*>(&ldsB[wave][buf][lo16][48 + hi * 8]);

    v8f acc = {};
    acc = __builtin_amdgcn_wmma_f32_16x16x32_f16(false, aLo.v, false, bLo.v,
                                                 (short)0, acc, false, false);
    acc = __builtin_amdgcn_wmma_f32_16x16x32_f16(false, aHi.v, false, bHi.v,
                                                 (short)0, acc, false, false);

    const int mv = mp[(wave << 7) + (c << 4) + lo16];
    #pragma unroll
    for (int r = 0; r < 8; ++r)
      acc[r] = (mv == 0) ? -1.0e9f : acc[r] * 0.125f;   // 1/sqrt(64)
    sc[c] = acc;

    if (c < 7) storeQ(buf ^ 1);       // drain next chunk's stores under WMMA
  }

  // ---- softmax over t: wave-local shuffle + cross-wave LDS reduction ----
  float rmax[8], rsum[8];
  #pragma unroll
  for (int r = 0; r < 8; ++r) {
    float v = sc[0][r];
    #pragma unroll
    for (int c = 1; c < 8; ++c) v = fmaxf(v, sc[c][r]);
    #pragma unroll
    for (int off = 1; off < 16; off <<= 1)
      v = fmaxf(v, __shfl_xor(v, off, 32));
    rmax[r] = v;
  }
  if (lo16 == 0) {
    #pragma unroll
    for (int r = 0; r < 8; ++r) ldsRed[0][wave][hi * 8 + r] = rmax[r];
  }
  __syncthreads();
  #pragma unroll
  for (int r = 0; r < 8; ++r) {
    float v = ldsRed[0][0][hi * 8 + r];
    #pragma unroll
    for (int w2 = 1; w2 < 8; ++w2) v = fmaxf(v, ldsRed[0][w2][hi * 8 + r]);
    rmax[r] = v;
    rsum[r] = 0.0f;
  }
  #pragma unroll
  for (int c = 0; c < 8; ++c) {
    #pragma unroll
    for (int r = 0; r < 8; ++r) {
      const float e = __expf(sc[c][r] - rmax[r]);
      sc[c][r] = e;
      rsum[r] += e;
    }
  }
  #pragma unroll
  for (int r = 0; r < 8; ++r) {
    float v = rsum[r];
    #pragma unroll
    for (int off = 1; off < 16; off <<= 1) v += __shfl_xor(v, off, 32);
    rsum[r] = v;
  }
  if (lo16 == 0) {
    #pragma unroll
    for (int r = 0; r < 8; ++r) ldsRed[1][wave][hi * 8 + r] = rsum[r];
  }
  __syncthreads();
  #pragma unroll
  for (int r = 0; r < 8; ++r) {
    float v = 0.0f;
    #pragma unroll
    for (int w2 = 0; w2 < 8; ++w2) v += ldsRed[1][w2][hi * 8 + r];
    rsum[r] = 1.0f / v;
  }

  // ---- write attn[bh][s0+row][t] (64B-coalesced per 16-lane group) ----
  float* arow = attn + ((size_t)bh * SEQ + s0) * SEQ;
  #pragma unroll
  for (int c = 0; c < 8; ++c) {
    const int t = (wave << 7) + (c << 4) + lo16;
    #pragma unroll
    for (int r = 0; r < 8; ++r)
      arow[(size_t)(hi * 8 + r) * SEQ + t] = sc[c][r] * rsum[r];
  }
}

// ===========================================================================
// Kernel 2: ctx[b,t,h*64+d] = sum_s V[d,s]*attn[s,t].
// Block: 256 threads (8 waves) -> 64(d) x 128(t) tile; K-loop over s, 32/iter,
// double-buffered staging; output transpose buffer unioned with staging LDS.
// ===========================================================================
__global__ __launch_bounds__(256)
void attn_av_kernel(const float* __restrict__ Vg,
                    const float* __restrict__ attn,
                    float*       __restrict__ ctx)
{
  __shared__ union SmemK2 {
    struct {
      _Float16 V[2][64][40];     // [buf][d][s]  (80B rows)
      _Float16 P[2][128][40];    // [buf][t][s]
    } st;
    float O[128][68];            // output transpose [t][d] (272B rows)
  } sm;

  const int bh = blockIdx.y;
  const int b  = bh / NHEAD;
  const int h  = bh % NHEAD;
  const int tb = blockIdx.x << 7;

  const float* Vp = Vg   + (size_t)bh * DH * SEQ;
  const float* Ap = attn + (size_t)bh * SEQ * SEQ;

  const int tid  = threadIdx.x;
  const int wave = tid >> 5;
  const int lane = tid & 31;
  const int lo16 = lane & 15;
  const int hi   = lane >> 4;

  // staging assignments
  const int dV    = tid >> 2;          // 0..63   (V tile)
  const int cV    = (tid & 3) << 3;    // 0,8,16,24
  const int tcol0 = (tid & 31) << 2;   // 0..124  (attn tile, 4 t-cols)
  const int rb    = (tid >> 5) << 2;   // 0..28   (4 consecutive s-rows)

  float4 vreg[2], preg[4];
  auto loadRegs = [&](int i) {
    const int sc0 = i << 5;
    vreg[0] = *reinterpret_cast<const float4*>(Vp + (size_t)dV * SEQ + sc0 + cV);
    vreg[1] = *reinterpret_cast<const float4*>(Vp + (size_t)dV * SEQ + sc0 + cV + 4);
    #pragma unroll
    for (int j = 0; j < 4; ++j)   // 512B fully coalesced rows of attn
      preg[j] = *reinterpret_cast<const float4*>(
          Ap + (size_t)(sc0 + rb + j) * SEQ + tb + tcol0);
  };
  auto storeLds = [&](int buf) {
    Pack8 pv;
    pv.h[0] = (_Float16)vreg[0].x;  pv.h[1] = (_Float16)vreg[0].y;
    pv.h[2] = (_Float16)vreg[0].z;  pv.h[3] = (_Float16)vreg[0].w;
    pv.h[4] = (_Float16)vreg[1].x;  pv.h[5] = (_Float16)vreg[1].y;
    pv.h[6] = (_Float16)vreg[1].z;  pv.h[7] = (_Float16)vreg[1].w;
    *reinterpret_cast<uint4*>(&sm.st.V[buf][dV][cV]) = pv.u;   // one b128
    // 4x4 in-register transpose -> 4 packed b64 stores (s-contiguous)
    Pack4 p0, p1, p2, p3;
    p0.h[0] = (_Float16)preg[0].x; p0.h[1] = (_Float16)preg[1].x;
    p0.h[2] = (_Float16)preg[2].x; p0.h[3] = (_Float16)preg[3].x;
    p1.h[0] = (_Float16)preg[0].y; p1.h[1] = (_Float16)preg[1].y;
    p1.h[2] = (_Float16)preg[2].y; p1.h[3] = (_Float16)preg[3].y;
    p2.h[0] = (_Float16)preg[0].z; p2.h[1] = (_Float16)preg[1].z;
    p2.h[2] = (_Float16)preg[2].z; p2.h[3] = (_Float16)preg[3].z;
    p3.h[0] = (_Float16)preg[0].w; p3.h[1] = (_Float16)preg[1].w;
    p3.h[2] = (_Float16)preg[2].w; p3.h[3] = (_Float16)preg[3].w;
    *reinterpret_cast<uint2*>(&sm.st.P[buf][tcol0 + 0][rb]) = p0.u;
    *reinterpret_cast<uint2*>(&sm.st.P[buf][tcol0 + 1][rb]) = p1.u;
    *reinterpret_cast<uint2*>(&sm.st.P[buf][tcol0 + 2][rb]) = p2.u;
    *reinterpret_cast<uint2*>(&sm.st.P[buf][tcol0 + 3][rb]) = p3.u;
  };

  const v8f z = {};
  v8f acc0 = z, acc1 = z, acc2 = z, acc3 = z;
  const int trow = (wave << 4) + lo16;

  loadRegs(0);
  storeLds(0);
  __syncthreads();

  for (int i = 0; i < 32; ++i) {
    const int buf = i & 1;
    if (i < 31) loadRegs(i + 1);      // issue next chunk's globals early

    V16 bP;
    bP.u[0] = *reinterpret_cast<const uint4*>(&sm.st.P[buf][trow][hi * 8]);
    bP.u[1] = *reinterpret_cast<const uint4*>(&sm.st.P[buf][trow][16 + hi * 8]);

    V16 aV;
    aV.u[0] = *reinterpret_cast<const uint4*>(&sm.st.V[buf][lo16][hi * 8]);
    aV.u[1] = *reinterpret_cast<const uint4*>(&sm.st.V[buf][lo16][16 + hi * 8]);
    acc0 = __builtin_amdgcn_wmma_f32_16x16x32_f16(false, aV.v, false, bP.v,
                                                  (short)0, acc0, false, false);
    aV.u[0] = *reinterpret_cast<const uint4*>(&sm.st.V[buf][16 + lo16][hi * 8]);
    aV.u[1] = *reinterpret_cast<const uint4*>(&sm.st.V[buf][16 + lo16][16 + hi * 8]);
    acc1 = __builtin_amdgcn_wmma_f32_16x16x32_f16(false, aV.v, false, bP.v,
                                                  (short)0, acc1, false, false);
    aV.u[0] = *reinterpret_cast<const uint4*>(&sm.st.V[buf][32 + lo16][hi * 8]);
    aV.u[1] = *reinterpret_cast<const uint4*>(&sm.st.V[buf][32 + lo16][16 + hi * 8]);
    acc2 = __builtin_amdgcn_wmma_f32_16x16x32_f16(false, aV.v, false, bP.v,
                                                  (short)0, acc2, false, false);
    aV.u[0] = *reinterpret_cast<const uint4*>(&sm.st.V[buf][48 + lo16][hi * 8]);
    aV.u[1] = *reinterpret_cast<const uint4*>(&sm.st.V[buf][48 + lo16][16 + hi * 8]);
    acc3 = __builtin_amdgcn_wmma_f32_16x16x32_f16(false, aV.v, false, bP.v,
                                                  (short)0, acc3, false, false);

    if (i < 31) storeLds(buf ^ 1);    // drain stores under WMMA
    __syncthreads();                  // one barrier per iteration
  }

  // ---- epilogue: transpose D tiles via LDS, then coalesced float4 stores ----
  #pragma unroll
  for (int r = 0; r < 8; ++r) {
    sm.O[trow][ 0 + hi * 8 + r] = acc0[r];
    sm.O[trow][16 + hi * 8 + r] = acc1[r];
    sm.O[trow][32 + hi * 8 + r] = acc2[r];
    sm.O[trow][48 + hi * 8 + r] = acc3[r];
  }
  __syncthreads();
  #pragma unroll
  for (int k2 = 0; k2 < 8; ++k2) {
    const int tl = (k2 << 4) + (tid >> 4);
    const int c0 = (tid & 15) << 2;
    float4 o;
    o.x = sm.O[tl][c0 + 0];
    o.y = sm.O[tl][c0 + 1];
    o.z = sm.O[tl][c0 + 2];
    o.w = sm.O[tl][c0 + 3];
    *reinterpret_cast<float4*>(
        ctx + ((size_t)b * SEQ + tb + tl) * (NHEAD * DH) + h * DH + c0) = o;
  }
}

// ===========================================================================
extern "C" void kernel_launch(void* const* d_in, const int* in_sizes, int n_in,
                              void* d_out, int out_size, void* d_ws, size_t ws_size,
                              hipStream_t stream) {
  (void)in_sizes; (void)n_in; (void)out_size; (void)d_ws; (void)ws_size;
  const float* Q    = (const float*)d_in[0];
  const float* K    = (const float*)d_in[1];
  const float* V    = (const float*)d_in[2];
  const int*   mask = (const int*)d_in[3];

  float* ctx  = (float*)d_out;               // (B, S, H*D)
  float* attn = (float*)d_out + CTX_ELEMS;   // (B, H, S, S)

  dim3 blk(256);
  dim3 g1(SEQ / 16, NBATCH * NHEAD);         // 64 x 96
  attn_qk_softmax_kernel<<<g1, blk, 0, stream>>>(Q, K, mask, attn);

  dim3 g2(SEQ / 128, NBATCH * NHEAD);        // 8 x 96
  attn_av_kernel<<<g2, blk, 0, stream>>>(V, attn, ctx);
}
